// AtomicLinear_16844861735162
// MI455X (gfx1250) — compile-verified, matrix operand order checked by reading, not run on hardware
//
#include <hip/hip_runtime.h>
#include <stdint.h>

typedef __attribute__((ext_vector_type(16))) __bf16      v16bf;
typedef __attribute__((ext_vector_type(8)))  float        v8f;
typedef __attribute__((ext_vector_type(8)))  unsigned int v8u;
typedef __attribute__((ext_vector_type(4)))  int          v4i;

#define TM   128   // block tile M (batch rows)
#define TN   128   // block tile N (output features)
#define TK   64    // K depth staged per buffer (2 WMMA sub-steps of 32)
#define LSTR 72    // padded LDS row stride in bf16 elems (64 + 8 -> 144B rows)

#if defined(__AMDGCN__) && __has_builtin(__builtin_amdgcn_global_load_async_to_lds_b128)
#define HAS_ASYNC_LDS 1
typedef __attribute__((address_space(1))) v4i* gv4_t;   // global int4*
typedef __attribute__((address_space(3))) v4i* lv4_t;   // LDS int4*
#else
#define HAS_ASYNC_LDS 0
#endif

__device__ __forceinline__ void wait_async0() {
#if HAS_ASYNC_LDS
#if __has_builtin(__builtin_amdgcn_s_wait_asynccnt)
  __builtin_amdgcn_s_wait_asynccnt(0);
#else
  asm volatile("s_wait_asynccnt 0" ::: "memory");
#endif
#endif
}

// f32 pair -> packed bf16 (RNE): bias-add then v_perm_b32 grabs the two hi16s.
__device__ __forceinline__ unsigned pk2_bf16(float a, float b) {
  unsigned ua = __builtin_bit_cast(unsigned, a);
  unsigned ub = __builtin_bit_cast(unsigned, b);
  ua += 0x7FFFu + ((ua >> 16) & 1u);
  ub += 0x7FFFu + ((ub >> 16) & 1u);
  return __builtin_amdgcn_perm(ub, ua, 0x07060302u);  // {hi16(b), hi16(a)}
}

// ---------------- pass 1: f32 -> bf16 bulk convert (memory bound) ------------
__global__ __launch_bounds__(256)
void cvt_f32_bf16(const float4* __restrict__ in, uint2* __restrict__ out, int n4) {
  int i = blockIdx.x * 256 + threadIdx.x;
  const int stride = gridDim.x * 256;
  for (; i < n4; i += stride) {
    const float4 v = in[i];
    uint2 p;
    p.x = pk2_bf16(v.x, v.y);
    p.y = pk2_bf16(v.z, v.w);
    out[i] = p;
  }
}

// Build a lane's 16x32 bf16 WMMA fragment from an LDS row (two ds_load_b128).
__device__ __forceinline__ v16bf load_frag(const unsigned short* p, unsigned kh) {
  const uint4 lo = *(const uint4*)(p + kh);        // K = kh .. kh+7
  const uint4 hi = *(const uint4*)(p + 16 + kh);   // K = 16+kh .. 16+kh+7
  v8u t;
  t[0] = lo.x; t[1] = lo.y; t[2] = lo.z; t[3] = lo.w;
  t[4] = hi.x; t[5] = hi.y; t[6] = hi.z; t[7] = hi.w;
  return __builtin_bit_cast(v16bf, t);
}

// ---------------- pass 2: bf16 WMMA GEMM ------------------------------------
// F32IN=true: fused fallback (f32 inputs, convert on the LDS-store path).
// F32IN=false: inputs pre-converted bf16; tiles staged with async global->LDS.
template <bool F32IN>
__global__ __launch_bounds__(256, 2)
void gemm_wmma(const void* __restrict__ Xp, const void* __restrict__ Wp,
               const float* __restrict__ bias, float* __restrict__ out,
               int K, int N) {
  __shared__ __align__(16) unsigned short ldsA[2][TM * LSTR];
  __shared__ __align__(16) unsigned short ldsB[2][TN * LSTR];

  const unsigned tid  = threadIdx.x;
  const unsigned lane = tid & 31u;
  const unsigned wave = tid >> 5;        // 8 waves: 2 (M) x 4 (N)
  const unsigned wm   = wave & 1u;       // 64-row slab
  const unsigned wn   = wave >> 1;       // 32-col slab
  const unsigned lm   = lane & 15u;
  const unsigned kh   = (lane >> 4) << 3;  // K-half select (0 or 8)

  const unsigned rowBase = blockIdx.y * TM;
  const unsigned colBase = blockIdx.x * TN;
  const unsigned uK = (unsigned)K, uN = (unsigned)N;
  const int nKT = K / TK;

  // staging for F32IN path: 128x64 f32 tile = 2048 float4, 8 per thread
  float4 aF[8], bF[8];
  const unsigned rb16 = tid >> 4, c4 = tid & 15u;   // row base, float4 col
  // staging for bf16 non-async fallback: 128x64 bf16 tile = 1024 x 16B
  uint4 aU[4], bU[4];
  const unsigned rb8 = tid >> 3, cc = tid & 7u;     // row base, 16B-chunk col

  auto load_tile = [&](int kt, int dbuf) {
    if constexpr (F32IN) {
      const float* X = (const float*)Xp;
      const float* W = (const float*)Wp;
      const unsigned colOff = (unsigned)kt * TK + c4 * 4u;
#pragma unroll
      for (int i = 0; i < 8; ++i) {
        const unsigned r = rb16 + i * 16u;
        aF[i] = *(const float4*)(X + (rowBase + r) * uK + colOff);
        bF[i] = *(const float4*)(W + (colBase + r) * uK + colOff);
      }
    } else {
      const unsigned short* Xb = (const unsigned short*)Xp;
      const unsigned short* Wb = (const unsigned short*)Wp;
      const unsigned colOff = (unsigned)kt * TK + cc * 8u;
#pragma unroll
      for (int i = 0; i < 4; ++i) {
        const unsigned r = rb8 + i * 32u;
        const unsigned short* ga = Xb + (rowBase + r) * uK + colOff;
        const unsigned short* gb = Wb + (colBase + r) * uK + colOff;
        unsigned short* la = &ldsA[dbuf][r * LSTR + cc * 8u];
        unsigned short* lb = &ldsB[dbuf][r * LSTR + cc * 8u];
#if HAS_ASYNC_LDS
        __builtin_amdgcn_global_load_async_to_lds_b128(
            (gv4_t)(unsigned long long)ga, (lv4_t)la, 0, 0);
        __builtin_amdgcn_global_load_async_to_lds_b128(
            (gv4_t)(unsigned long long)gb, (lv4_t)lb, 0, 0);
#else
        aU[i] = *(const uint4*)ga;
        bU[i] = *(const uint4*)gb;
#endif
      }
    }
  };

  auto store_tile = [&](int dbuf) {
    if constexpr (F32IN) {
#pragma unroll
      for (int i = 0; i < 8; ++i) {
        const unsigned r = rb16 + i * 16u;
        uint2 pa, pb;
        pa.x = pk2_bf16(aF[i].x, aF[i].y); pa.y = pk2_bf16(aF[i].z, aF[i].w);
        pb.x = pk2_bf16(bF[i].x, bF[i].y); pb.y = pk2_bf16(bF[i].z, bF[i].w);
        *(uint2*)&ldsA[dbuf][r * LSTR + c4 * 4u] = pa;
        *(uint2*)&ldsB[dbuf][r * LSTR + c4 * 4u] = pb;
      }
    } else {
#if !HAS_ASYNC_LDS
#pragma unroll
      for (int i = 0; i < 4; ++i) {
        const unsigned r = rb8 + i * 32u;
        *(uint4*)&ldsA[dbuf][r * LSTR + cc * 8u] = aU[i];
        *(uint4*)&ldsB[dbuf][r * LSTR + cc * 8u] = bU[i];
      }
#endif
    }
  };

  auto sync_tile = [&]() {
    if constexpr (!F32IN) wait_async0();  // drain this wave's async copies
    __syncthreads();
  };

  v8f acc[4][2];
  const v8f vzero = {0.f, 0.f, 0.f, 0.f, 0.f, 0.f, 0.f, 0.f};
#pragma unroll
  for (int mi = 0; mi < 4; ++mi)
#pragma unroll
    for (int ni = 0; ni < 2; ++ni) acc[mi][ni] = vzero;

  int buf = 0;
  load_tile(0, 0);
  store_tile(0);
  sync_tile();

  for (int kt = 0; kt < nKT; ++kt) {
    if (kt + 1 < nKT) load_tile(kt + 1, buf ^ 1);  // overlaps WMMA below

    // two 32-deep WMMA sub-steps per staged 64-deep tile
#pragma unroll
    for (int ks = 0; ks < TK / 32; ++ks) {
      const unsigned kcol = ks * 32u;
      v16bf afrag[4], bfrag[2];
#pragma unroll
      for (int mi = 0; mi < 4; ++mi)
        afrag[mi] =
            load_frag(&ldsA[buf][(wm * 64u + mi * 16u + lm) * LSTR + kcol], kh);
#pragma unroll
      for (int ni = 0; ni < 2; ++ni)
        bfrag[ni] =
            load_frag(&ldsB[buf][(wn * 32u + ni * 16u + lm) * LSTR + kcol], kh);

#pragma unroll
      for (int mi = 0; mi < 4; ++mi)
#pragma unroll
        for (int ni = 0; ni < 2; ++ni)
          acc[mi][ni] = __builtin_amdgcn_wmma_f32_16x16x32_bf16(
              false, afrag[mi], false, bfrag[ni],
              (short)0, acc[mi][ni], false, false);
    }

    if (kt + 1 < nKT) {
      store_tile(buf ^ 1);
      sync_tile();
      buf ^= 1;
    }
  }

  // Epilogue: C layout = 8 VGPRs; VGPR r -> M=r (lanes 0-15) / r+8 (16-31).
  // Pure 32-bit indexing (out has < 2^31 elements) to avoid 64-bit mul chains.
#pragma unroll
  for (int ni = 0; ni < 2; ++ni) {
    const unsigned col = colBase + wn * 32u + ni * 16u + lm;
    const float bv = bias[col];
#pragma unroll
    for (int mi = 0; mi < 4; ++mi) {
      const unsigned mrow = rowBase + wm * 64u + mi * 16u + (lane >> 4) * 8u;
#pragma unroll
      for (int r = 0; r < 8; ++r)
        out[(mrow + (unsigned)r) * uN + col] = acc[mi][ni][r] + bv;
    }
  }
}

extern "C" void kernel_launch(void* const* d_in, const int* in_sizes, int n_in,
                              void* d_out, int out_size, void* d_ws, size_t ws_size,
                              hipStream_t stream) {
  (void)n_in; (void)out_size;
  const float* x  = (const float*)d_in[0];
  const float* w  = (const float*)d_in[1];
  const float* bs = (const float*)d_in[2];
  float* out = (float*)d_out;

  const int OUT = in_sizes[2];        // 2048
  const int IN  = in_sizes[1] / OUT;  // 2048
  const int B   = in_sizes[0] / IN;   // 8192
  dim3 grid(OUT / TN, B / TM);

  const size_t needX = (size_t)B * IN * sizeof(unsigned short);
  const size_t needW = (size_t)OUT * IN * sizeof(unsigned short);

  if (ws_size >= needX + needW) {
    // two-pass: bulk convert to bf16 in workspace, then pure-bf16 WMMA GEMM
    unsigned short* xb = (unsigned short*)d_ws;
    unsigned short* wb = xb + (size_t)B * IN;
    const int n4x = (B * IN) / 4;
    const int n4w = (OUT * IN) / 4;
    cvt_f32_bf16<<<4096, 256, 0, stream>>>((const float4*)x, (uint2*)xb, n4x);
    cvt_f32_bf16<<<4096, 256, 0, stream>>>((const float4*)w, (uint2*)wb, n4w);
    gemm_wmma<false><<<grid, 256, 0, stream>>>(xb, wb, bs, out, IN, OUT);
  } else {
    // fused fallback: convert on the global->LDS path
    gemm_wmma<true><<<grid, 256, 0, stream>>>(x, w, bs, out, IN, OUT);
  }
}
